// AttentionWithRelPos_66116726555238
// MI455X (gfx1250) — compile-verified, matrix operand order checked by reading, not run on hardware
//
#include <hip/hip_runtime.h>
#include <hip/hip_bf16.h>

typedef __attribute__((ext_vector_type(16))) _Float16 v16h;
typedef __attribute__((ext_vector_type(8)))  _Float16 v8h;
typedef __attribute__((ext_vector_type(8)))  float    v8f;

// Problem constants
#define BATCH   64
#define NSEQ    197
#define NPAD    224          // 7*32, covers 197
#define MFLAT   (BATCH * NPAD)   // 14336 = 224 * 64 -> exact 64-row tiling
#define DIM     768
#define HEADS   12
#define HD      64
#define R2      729          // 27*27
#define NWIN    196
#define ATT_SCALE 0.125f     // 64^-0.5

__device__ __forceinline__ v8f wmma_f16(v16h a, v16h b, v8f c) {
    return __builtin_amdgcn_wmma_f32_16x16x32_f16(
        false, a, false, b, (short)0, c, false, false);
}

// 16x32 f16 fragment. lane l: row = l&15, K = (l>>4)*8+j and +16.
__device__ __forceinline__ v16h frag_from_f16(const _Float16* __restrict__ p, int ld, int lane) {
    const int r = lane & 15, g = lane >> 4;
    const _Float16* p0 = p + (size_t)r * ld + g * 8;
    v8h lo = *(const v8h*)p0;
    v8h hi = *(const v8h*)(p0 + 16);
    v16h a;
#pragma unroll
    for (int j = 0; j < 8; ++j) { a[j] = lo[j]; a[8 + j] = hi[j]; }
    return a;
}

// ---------------------------------------------------------------------------
// Prep 1: weights fp32 -> f16 (one shot; removes cvt from GEMM inner loops)
// ---------------------------------------------------------------------------
__global__ __launch_bounds__(256) void convert_w_kernel(const float* __restrict__ wq,
                                                        const float* __restrict__ wp,
                                                        _Float16* __restrict__ wq16,
                                                        _Float16* __restrict__ wp16) {
    const size_t i = (size_t)blockIdx.x * 256 + threadIdx.x;
    if (i < (size_t)3 * DIM * DIM) wq16[i] = (_Float16)wq[i];
    if (i < (size_t)DIM * DIM)     wp16[i] = (_Float16)wp[i];
}

// ---------------------------------------------------------------------------
// Prep 2: x fp32 -> f16, zero-padded to [B][NPAD][DIM]
// ---------------------------------------------------------------------------
__global__ __launch_bounds__(256) void pad_x_kernel(const float* __restrict__ x,
                                                    _Float16* __restrict__ x16) {
    const size_t i = (size_t)blockIdx.x * 256 + threadIdx.x;
    if (i >= (size_t)BATCH * NPAD * DIM) return;
    const int d   = (int)(i % DIM);
    const int n   = (int)((i / DIM) % NPAD);
    const int b   = (int)(i / ((size_t)NPAD * DIM));
    x16[i] = (n < NSEQ) ? (_Float16)x[((size_t)b * NSEQ + n) * DIM + d] : (_Float16)0.f;
}

// ---------------------------------------------------------------------------
// Prep 3: masked rel-pos bias table [12][224][224] f32 (reused by all 64 batches)
// bias = -1e30 for padded keys; rel_pos gather for interior; 0 for cls row/col.
// ---------------------------------------------------------------------------
__global__ __launch_bounds__(256) void bias_kernel(const float* __restrict__ rel_pos,
                                                   const int* __restrict__ rpidx,
                                                   float* __restrict__ biasbuf) {
    const size_t i = (size_t)blockIdx.x * 256 + threadIdx.x;
    if (i >= (size_t)HEADS * NPAD * NPAD) return;
    const int ki = (int)(i % NPAD);
    const int qi = (int)((i / NPAD) % NPAD);
    const int h  = (int)(i / ((size_t)NPAD * NPAD));
    float val;
    if (ki >= NSEQ) val = -1.0e30f;
    else if (qi >= 1 && qi < NSEQ && ki >= 1)
        val = rel_pos[h * R2 + rpidx[(qi - 1) * NWIN + (ki - 1)]];
    else val = 0.0f;
    biasbuf[i] = val;
}

// ---------------------------------------------------------------------------
// Kernel 1: qkv GEMM, 64x64 tile per wave (16 WMMA / 16 b128-loads per K-step)
// M flattened over B*NPAD. grid (36, 224), block 32.
// ---------------------------------------------------------------------------
__global__ __launch_bounds__(32) void qkv_kernel(const _Float16* __restrict__ x16,
                                                 const _Float16* __restrict__ w16,
                                                 _Float16* __restrict__ qbuf,
                                                 _Float16* __restrict__ kbuf,
                                                 _Float16* __restrict__ vtbuf) {
    const int lane = threadIdx.x;
    const int r = lane & 15, g = lane >> 4;
    const int nt = blockIdx.x;            // 0..35 -> 64-wide output col group
    const int mt = blockIdx.y;            // 0..223 -> 64-row tile of flat M
    const int which = nt / 12;            // 0=q 1=k 2=v
    const int h     = nt % 12;
    const int j0    = nt * 64;

    const _Float16* abase = x16 + (size_t)mt * 64 * DIM;

    v8f acc[4][4];
#pragma unroll
    for (int rg = 0; rg < 4; ++rg)
#pragma unroll
        for (int c = 0; c < 4; ++c)
#pragma unroll
            for (int i = 0; i < 8; ++i) acc[rg][c][i] = 0.0f;

    for (int kk = 0; kk < DIM; kk += 32) {
        if (kk + 64 < DIM) __builtin_prefetch(abase + kk + 64, 0, 0);
        v16h af[4];
#pragma unroll
        for (int rg = 0; rg < 4; ++rg)
            af[rg] = frag_from_f16(abase + (size_t)rg * 16 * DIM + kk, DIM, lane);
#pragma unroll
        for (int c = 0; c < 4; ++c) {
            v16h bf = frag_from_f16(w16 + (size_t)(j0 + c * 16) * DIM + kk, DIM, lane);
#pragma unroll
            for (int rg = 0; rg < 4; ++rg)
                acc[rg][c] = wmma_f16(af[rg], bf, acc[rg][c]);
        }
    }

#pragma unroll
    for (int rg = 0; rg < 4; ++rg) {
        const int m0 = mt * 64 + rg * 16 + 8 * g;
#pragma unroll
        for (int c = 0; c < 4; ++c) {
            const int d = c * 16 + r;
#pragma unroll
            for (int i = 0; i < 8; ++i) {
                const int m = m0 + i;
                const int b = m / NPAD;
                const int n = m - b * NPAD;
                const size_t bh = (size_t)(b * HEADS + h);
                const float v = acc[rg][c][i];
                if (which == 0)      qbuf[(bh * NPAD + n) * HD + d] = (_Float16)(v * ATT_SCALE);
                else if (which == 1) kbuf[(bh * NPAD + n) * HD + d] = (_Float16)v;
                else                 vtbuf[(bh * HD + d) * NPAD + n] = (_Float16)v;
            }
        }
    }
}

// ---------------------------------------------------------------------------
// Kernel 2: attention, 16 query rows of one (b,h) per wave.
// grid (13, 12, 64), block 32.
// ---------------------------------------------------------------------------
__global__ __launch_bounds__(32) void attn_kernel(const _Float16* __restrict__ qbuf,
                                                  const _Float16* __restrict__ kbuf,
                                                  const _Float16* __restrict__ vtbuf,
                                                  const float* __restrict__ biasbuf,
                                                  _Float16* __restrict__ attn_out) {
    __shared__ float Sbuf[16][228];   // 228-dword stride -> conflict-free rows
    __shared__ v8h   Pbuf[16][29];    // 232 halves per row
    __shared__ float rinv[16];

    const int lane = threadIdx.x;
    const int r = lane & 15, g = lane >> 4;
    const int q0 = blockIdx.x * 16;
    const int h  = blockIdx.y;
    const int b  = blockIdx.z;
    const size_t bh = (size_t)(b * HEADS + h);

    const _Float16* qbase = qbuf + (bh * NPAD + q0) * HD;   // SCALE folded in
    const v16h qa0 = frag_from_f16(qbase,      HD, lane);
    const v16h qa1 = frag_from_f16(qbase + 32, HD, lane);
    const float* brow = biasbuf + (size_t)h * NPAD * NPAD;

    // ---- S = q.k^T + bias (mask baked into bias table) ----
    for (int kt = 0; kt < NPAD / 16; ++kt) {
        const _Float16* kbase = kbuf + (bh * NPAD + kt * 16) * HD;
        v16h kb0 = frag_from_f16(kbase,      HD, lane);
        v16h kb1 = frag_from_f16(kbase + 32, HD, lane);
        v8f s;
#pragma unroll
        for (int i = 0; i < 8; ++i) s[i] = 0.0f;
        s = wmma_f16(qa0, kb0, s);
        s = wmma_f16(qa1, kb1, s);

        const int ki = kt * 16 + r;
#pragma unroll
        for (int i = 0; i < 8; ++i) {
            const int qi = q0 + i + 8 * g;
            Sbuf[i + 8 * g][kt * 16 + r] = s[i] + brow[(size_t)qi * NPAD + ki];
        }
    }
    __syncthreads();

    // ---- fp32 softmax: lane owns row (l&15), half (l>>4) of 224 columns ----
    {
        const int row = r;
        const int c0  = g * 112;
        float m = -3.0e38f;
        for (int c = 0; c < 112; ++c) m = fmaxf(m, Sbuf[row][c0 + c]);
        m = fmaxf(m, __shfl_xor(m, 16, 32));
        float ssum = 0.0f;
        _Float16* Prow = (_Float16*)&Pbuf[row][0];
        for (int c = 0; c < 112; ++c) {
            const float e = __expf(Sbuf[row][c0 + c] - m);
            ssum += e;
            Prow[c0 + c] = (_Float16)e;       // unnormalized; 1/sum in epilogue
        }
        ssum += __shfl_xor(ssum, 16, 32);
        if (g == 0) rinv[row] = 1.0f / ssum;
    }
    __syncthreads();

    // ---- O = P @ v ----
    v8f o[4];
#pragma unroll
    for (int c = 0; c < 4; ++c)
#pragma unroll
        for (int i = 0; i < 8; ++i) o[c][i] = 0.0f;

    for (int pt = 0; pt < NPAD / 32; ++pt) {
        v16h pa;
        {
            v8h lo = Pbuf[r][pt * 4 + g];
            v8h hi = Pbuf[r][pt * 4 + 2 + g];
#pragma unroll
            for (int j = 0; j < 8; ++j) { pa[j] = lo[j]; pa[8 + j] = hi[j]; }
        }
#pragma unroll
        for (int c = 0; c < 4; ++c) {
            const _Float16* vbase = vtbuf + (bh * HD + c * 16) * NPAD + pt * 32;
            v16h vb = frag_from_f16(vbase, NPAD, lane);
            o[c] = wmma_f16(pa, vb, o[c]);
        }
    }

#pragma unroll
    for (int c = 0; c < 4; ++c) {
#pragma unroll
        for (int i = 0; i < 8; ++i) {
            const int nrow  = q0 + i + 8 * g;
            const float val = o[c][i] * rinv[i + 8 * g];
            attn_out[((size_t)b * NPAD + nrow) * DIM + h * HD + c * 16 + r] = (_Float16)val;
        }
    }
}

// ---------------------------------------------------------------------------
// Kernel 3: out = attn_out @ w_proj^T + b_proj, 64x64 tile per wave.
// grid (12, 224), block 32.
// ---------------------------------------------------------------------------
__global__ __launch_bounds__(32) void proj_kernel(const _Float16* __restrict__ attn_out,
                                                  const _Float16* __restrict__ wp16,
                                                  const float* __restrict__ b_proj,
                                                  float* __restrict__ out) {
    const int lane = threadIdx.x;
    const int r = lane & 15, g = lane >> 4;
    const int j0 = blockIdx.x * 64;
    const int mt = blockIdx.y;

    const _Float16* abase = attn_out + (size_t)mt * 64 * DIM;

    v8f acc[4][4];
#pragma unroll
    for (int rg = 0; rg < 4; ++rg)
#pragma unroll
        for (int c = 0; c < 4; ++c)
#pragma unroll
            for (int i = 0; i < 8; ++i) acc[rg][c][i] = 0.0f;

    for (int kk = 0; kk < DIM; kk += 32) {
        if (kk + 64 < DIM) __builtin_prefetch(abase + kk + 64, 0, 0);
        v16h af[4];
#pragma unroll
        for (int rg = 0; rg < 4; ++rg)
            af[rg] = frag_from_f16(abase + (size_t)rg * 16 * DIM + kk, DIM, lane);
#pragma unroll
        for (int c = 0; c < 4; ++c) {
            v16h bf = frag_from_f16(wp16 + (size_t)(j0 + c * 16) * DIM + kk, DIM, lane);
#pragma unroll
            for (int rg = 0; rg < 4; ++rg)
                acc[rg][c] = wmma_f16(af[rg], bf, acc[rg][c]);
        }
    }

#pragma unroll
    for (int rg = 0; rg < 4; ++rg) {
        const int m0 = mt * 64 + rg * 16 + 8 * g;
#pragma unroll
        for (int c = 0; c < 4; ++c) {
            const float bj = b_proj[j0 + c * 16 + r];
#pragma unroll
            for (int i = 0; i < 8; ++i) {
                const int m = m0 + i;
                const int b = m / NPAD;
                const int n = m - b * NPAD;
                if (n < NSEQ)
                    out[((size_t)b * NSEQ + n) * DIM + j0 + c * 16 + r] = acc[rg][c][i] + bj;
            }
        }
    }
}

// ---------------------------------------------------------------------------
extern "C" void kernel_launch(void* const* d_in, const int* in_sizes, int n_in,
                              void* d_out, int out_size, void* d_ws, size_t ws_size,
                              hipStream_t stream) {
    const float* x       = (const float*)d_in[0];
    const float* w_qkv   = (const float*)d_in[1];
    const float* w_proj  = (const float*)d_in[2];
    const float* b_proj  = (const float*)d_in[3];
    const float* rel_pos = (const float*)d_in[4];
    const int*   rpidx   = (const int*)d_in[5];
    float* out = (float*)d_out;

    char* ws = (char*)d_ws;
    const size_t QKV_BYTES  = (size_t)BATCH * HEADS * NPAD * HD * sizeof(_Float16); // 22 MB
    const size_t X16_BYTES  = (size_t)BATCH * NPAD * DIM * sizeof(_Float16);        // 22 MB
    const size_t WQ_BYTES   = (size_t)3 * DIM * DIM * sizeof(_Float16);
    const size_t WP_BYTES   = (size_t)DIM * DIM * sizeof(_Float16);
    const size_t BIAS_BYTES = (size_t)HEADS * NPAD * NPAD * sizeof(float);

    size_t off = 0;
    _Float16* qb   = (_Float16*)(ws + off); off += QKV_BYTES;
    _Float16* kb   = (_Float16*)(ws + off); off += QKV_BYTES;
    _Float16* vt   = (_Float16*)(ws + off); off += QKV_BYTES;
    _Float16* ao   = (_Float16*)(ws + off); off += QKV_BYTES;  // [B][NPAD][DIM] f16
    _Float16* x16  = (_Float16*)(ws + off); off += X16_BYTES;
    _Float16* wq16 = (_Float16*)(ws + off); off += WQ_BYTES;
    _Float16* wp16 = (_Float16*)(ws + off); off += WP_BYTES;
    float*    bias = (float*)   (ws + off); off += BIAS_BYTES;
    (void)off; (void)ws_size;

    const int nW = (3 * DIM * DIM + 255) / 256;
    const int nX = (int)(((size_t)BATCH * NPAD * DIM + 255) / 256);
    const int nB = (HEADS * NPAD * NPAD + 255) / 256;

    convert_w_kernel<<<nW, 256, 0, stream>>>(w_qkv, w_proj, wq16, wp16);
    pad_x_kernel    <<<nX, 256, 0, stream>>>(x, x16);
    bias_kernel     <<<nB, 256, 0, stream>>>(rel_pos, rpidx, bias);

    qkv_kernel <<<dim3(36, MFLAT / 64), 32, 0, stream>>>(x16, wq16, qb, kb, vt);
    attn_kernel<<<dim3(13, HEADS, BATCH), 32, 0, stream>>>(qb, kb, vt, bias, ao);
    proj_kernel<<<dim3(12, MFLAT / 64), 32, 0, stream>>>(ao, wp16, b_proj, out);
}
